// Net_44942537786161
// MI455X (gfx1250) — matrix operand order repaired and checked by reading, not compiled, not run on hardware
//
#include <hip/hip_runtime.h>
#include <hip/hip_bf16.h>

typedef __attribute__((ext_vector_type(16))) _Float16 v16h;
typedef __attribute__((ext_vector_type(8)))  _Float16 v8h;
typedef __attribute__((ext_vector_type(8)))  float    v8f;

// ---------- helpers: order-preserving float<->uint for atomicMax on f32 ----------
static __device__ __forceinline__ unsigned ord_enc(float f) {
  unsigned b = __float_as_uint(f);
  return (b & 0x80000000u) ? ~b : (b | 0x80000000u);
}
static __device__ __forceinline__ float ord_dec(unsigned u) {
  unsigned b = (u & 0x80000000u) ? (u & 0x7fffffffu) : ~u;
  return __uint_as_float(b);
}

// ---------- generic utility kernels ----------
__global__ void k_f32_to_f16(const float* __restrict__ in, _Float16* __restrict__ out, long long n) {
  long long i = (long long)blockIdx.x * blockDim.x + threadIdx.x;
  if (i < n) out[i] = (_Float16)in[i];
}

__global__ void k_zero_b32(unsigned* __restrict__ p, long long n) {
  long long i = (long long)blockIdx.x * blockDim.x + threadIdx.x;
  if (i < n) p[i] = 0u;
}

// ---------- pack W (f32, KxNC row-major) into WMMA B-fragment lane order ----------
// Fragment (tileN tn, k-block kb): lane holds col = tn*16+(lane&15),
// 16 contiguous halves = K = kb*32 + (lane>>4)*16 + i.  Stored as consecutive v16h:
// P[((tn*KB + kb)*32 + lane)] = those 16 halves.
template <int K, int NC>
__global__ void k_pack_B(const float* __restrict__ W, _Float16* __restrict__ P) {
  const int KB = K / 32, ntn = NC / 16;
  int t = blockIdx.x * blockDim.x + threadIdx.x;
  if (t >= ntn * KB * 32) return;
  int lane = t & 31;
  int kb   = (t >> 5) % KB;
  int tn   = (t >> 5) / KB;
  int col  = tn * 16 + (lane & 15);
  int k0   = kb * 32 + (lane >> 4) * 16;
  _Float16* p = P + (size_t)t * 16;
#pragma unroll
  for (int i = 0; i < 16; ++i) p[i] = (_Float16)W[(size_t)(k0 + i) * NC + col];
}

// ---------- WMMA GEMM: C[M,NC] = A[M,K](f16) x Bpack(f16 fragments), f32 accum ----------
// One wave per 16x16 C tile. 50000 % 16 == 0 so no row padding needed.
template <int K, int NC>
__global__ void k_gemm_wmma(const _Float16* __restrict__ A, const v16h* __restrict__ Bp,
                            float* __restrict__ C, int M) {
  const int ntn = NC / 16, KB = K / 32;
  int wave  = (int)((blockIdx.x * (long long)blockDim.x + threadIdx.x) >> 5);
  int lane  = threadIdx.x & 31;
  int tileM = wave / ntn;
  int tileN = wave % ntn;
  if (tileM * 16 >= M) return;   // whole wave exits together -> EXEC all 1s for WMMA

  const int l15 = lane & 15;
  const int g   = lane >> 4;     // half-wave group
  const int row = tileM * 16 + l15;

  v8f acc = {};
#pragma unroll 2
  for (int kb = 0; kb < KB; ++kb) {
    // A fragment (16x32 f16): lane holds row; halves 0..15 = K[kk+16g..+15]
    // (contiguous, matching packed B). 16B-aligned.
    const _Float16* ap = A + (size_t)row * K + kb * 32 + g * 16;
    v8h lo = *(const v8h*)ap;
    v8h hi = *(const v8h*)(ap + 8);
    v16h a = __builtin_shufflevector(lo, hi, 0, 1, 2, 3, 4, 5, 6, 7,
                                     8, 9, 10, 11, 12, 13, 14, 15);
    // B fragment: one 32B vector load from the pre-packed layout
    v16h b = Bp[(size_t)(tileN * KB + kb) * 32 + lane];

    acc = __builtin_amdgcn_wmma_f32_16x16x32_f16(
        /*neg_a=*/false, a, /*neg_b=*/false, b,
        /*c_mod=*/(short)0, acc, /*reuse_a=*/false, /*reuse_b=*/false);
  }
  // C/D layout: lanes 0-15 -> M = r, lanes 16-31 -> M = 8 + r
  float* cp = C + (size_t)(tileM * 16 + g * 8) * NC + tileN * 16 + l15;
#pragma unroll
  for (int r = 0; r < 8; ++r) cp[(size_t)r * NC] = acc[r];
}

// ---------- per-node attention coefficients a_src/a_dst; also init m (ord-uint 0) and z ----------
__global__ void k_att_coeff(const float* __restrict__ xl,
                            const float* __restrict__ att_src, const float* __restrict__ att_dst,
                            float* __restrict__ a_src, float* __restrict__ a_dst,
                            unsigned* __restrict__ m, float* __restrict__ z,
                            int Nn, int H, int C) {
  int t = blockIdx.x * blockDim.x + threadIdx.x;
  if (t >= Nn * H) return;
  int h = t % H, n = t / H;
  const float* xr = xl + (size_t)n * H * C + (size_t)h * C;
  float s = 0.f, d = 0.f;
  for (int c = 0; c < C; ++c) {
    float v = xr[c];
    s += v * att_src[h * C + c];
    d += v * att_dst[h * C + c];
  }
  a_src[(size_t)n * H + h] = s;
  a_dst[(size_t)n * H + h] = d;
  m[(size_t)n * H + h] = 0u;     // ord-encoded global minimum
  z[(size_t)n * H + h] = 0.f;
}

// ---------- edge pass 1: segment max of leaky-relu logits (self-loops appended) ----------
__global__ void k_edge_max(const int* __restrict__ src, const int* __restrict__ dst,
                           int E, int Nn, int H,
                           const float* __restrict__ a_src, const float* __restrict__ a_dst,
                           unsigned* __restrict__ m) {
  long long t = (long long)blockIdx.x * blockDim.x + threadIdx.x;
  long long total = (long long)(E + Nn) * H;
  if (t >= total) return;
  int h = (int)(t % H);
  long long e = t / H;
  int s, d;
  if (e < E) { s = src[e]; d = dst[e]; } else { s = d = (int)(e - E); }
  float v = a_src[(size_t)s * H + h] + a_dst[(size_t)d * H + h];
  v = v > 0.f ? v : 0.2f * v;
  atomicMax(&m[(size_t)d * H + h], ord_enc(v));
}

// ---------- edge pass 2: z[dst,h] += exp(e - m[dst,h]) ----------
__global__ void k_edge_sum(const int* __restrict__ src, const int* __restrict__ dst,
                           int E, int Nn, int H,
                           const float* __restrict__ a_src, const float* __restrict__ a_dst,
                           const unsigned* __restrict__ m, float* __restrict__ z) {
  long long t = (long long)blockIdx.x * blockDim.x + threadIdx.x;
  long long total = (long long)(E + Nn) * H;
  if (t >= total) return;
  int h = (int)(t % H);
  long long e = t / H;
  int s, d;
  if (e < E) { s = src[e]; d = dst[e]; } else { s = d = (int)(e - E); }
  float v = a_src[(size_t)s * H + h] + a_dst[(size_t)d * H + h];
  v = v > 0.f ? v : 0.2f * v;
  float p = __expf(v - ord_dec(m[(size_t)d * H + h]));
  atomicAdd(&z[(size_t)d * H + h], p);
}

// ---------- edge pass 3: agg[dst, h, :] += alpha * xl[src, h, :] ----------
__global__ void k_edge_agg(const int* __restrict__ src, const int* __restrict__ dst,
                           int E, int Nn, int H, int C,
                           const float* __restrict__ a_src, const float* __restrict__ a_dst,
                           const unsigned* __restrict__ m, const float* __restrict__ z,
                           const float* __restrict__ xl, float* __restrict__ agg) {
  long long t = (long long)blockIdx.x * blockDim.x + threadIdx.x;
  long long total = (long long)(E + Nn) * H;
  if (t >= total) return;
  int h = (int)(t % H);
  long long e = t / H;
  int s, d;
  if (e < E) { s = src[e]; d = dst[e]; } else { s = d = (int)(e - E); }
  float v = a_src[(size_t)s * H + h] + a_dst[(size_t)d * H + h];
  v = v > 0.f ? v : 0.2f * v;
  float mm = ord_dec(m[(size_t)d * H + h]);
  float zz = z[(size_t)d * H + h];
  float alpha = __expf(v - mm) / (zz + 1e-16f);
  const float* xr = xl + ((size_t)s * H + h) * C;
  float* ar = agg + ((size_t)d * H + h) * C;
  __builtin_prefetch(xr, 0, 0);    // global_prefetch_b8 into L2-resident row
  for (int c = 0; c < C; ++c) atomicAdd(&ar[c], xr[c] * alpha);
}

// ---------- layer-1 finalize: h = f16( elu(agg + bias) ) ----------
__global__ void k_final1(const float* __restrict__ agg, const float* __restrict__ bias,
                         _Float16* __restrict__ out, int Nn, int F) {
  long long i = (long long)blockIdx.x * blockDim.x + threadIdx.x;
  if (i >= (long long)Nn * F) return;
  int f = (int)(i % F);
  float v = agg[i] + bias[f];
  v = v > 0.f ? v : (__expf(v) - 1.0f);
  out[i] = (_Float16)v;
}

// ---------- layer-2 finalize fused with log_softmax over 128 features (wave per node) ----------
__global__ void k_logsoftmax128(const float* __restrict__ agg, const float* __restrict__ bias,
                                float* __restrict__ out, int Nn) {
  int wave = (int)((blockIdx.x * (long long)blockDim.x + threadIdx.x) >> 5);
  int lane = threadIdx.x & 31;
  if (wave >= Nn) return;
  const float* row = agg + (size_t)wave * 128;
  float v[4];
  float mx = -3.4e38f;
#pragma unroll
  for (int i = 0; i < 4; ++i) {
    int f = lane * 4 + i;
    v[i] = row[f] + bias[f];
    mx = fmaxf(mx, v[i]);
  }
#pragma unroll
  for (int off = 16; off > 0; off >>= 1) mx = fmaxf(mx, __shfl_xor(mx, off, 32));
  float s = 0.f;
#pragma unroll
  for (int i = 0; i < 4; ++i) s += __expf(v[i] - mx);
#pragma unroll
  for (int off = 16; off > 0; off >>= 1) s += __shfl_xor(s, off, 32);
  float lse = mx + __logf(s);
  float* orow = out + (size_t)wave * 128;
#pragma unroll
  for (int i = 0; i < 4; ++i) orow[lane * 4 + i] = v[i] - lse;
}

// =================================================================================
static inline int cdivll(long long a, int b) { return (int)((a + b - 1) / b); }

extern "C" void kernel_launch(void* const* d_in, const int* in_sizes, int n_in,
                              void* d_out, int out_size, void* d_ws, size_t ws_size,
                              hipStream_t stream) {
  const float* x   = (const float*)d_in[0];
  const int*   ei  = (const int*)d_in[1];
  const float* W1  = (const float*)d_in[2];
  const float* as1 = (const float*)d_in[3];
  const float* ad1 = (const float*)d_in[4];
  const float* b1  = (const float*)d_in[5];
  const float* W2  = (const float*)d_in[6];
  const float* as2 = (const float*)d_in[7];
  const float* ad2 = (const float*)d_in[8];
  const float* b2  = (const float*)d_in[9];

  const int Nn = in_sizes[0] / 256;       // 50000
  const int E  = in_sizes[1] / 2;         // 1600000
  const int* src = ei;
  const int* dst = ei + E;
  const int H1 = 8, C1 = 8, F1 = 64;
  const int H2 = 8, C2 = 16, F2 = 128;

  // ---- workspace layout (bytes); xl2 overlays xh, agg2 overlays xl1+agg1 ----
  char* ws = (char*)d_ws;
  size_t o = 0;
  _Float16* xh   = (_Float16*)(ws + o);                 // N x 256 f16  (dead after gemm1)
  float*    xl2  = (float*)(ws + o);    o += (size_t)Nn * 512;   // N x 128 f32 overlay
  _Float16* W1p  = (_Float16*)(ws + o); o += 32768;     // packed B fragments, 256x64
  _Float16* W2p  = (_Float16*)(ws + o); o += 16384;     // packed B fragments, 64x128
  float*    xl1  = (float*)(ws + o);                    // N x 64 f32
  float*    agg2 = (float*)(ws + o);    o += (size_t)Nn * 256;   // N x 128 overlay (part 1)
  float*    agg1 = (float*)(ws + o);    o += (size_t)Nn * 256;   // N x 64, overlay part 2
  float*    a1s  = (float*)(ws + o);    o += (size_t)Nn * 32;
  float*    a1d  = (float*)(ws + o);    o += (size_t)Nn * 32;
  unsigned* m1   = (unsigned*)(ws + o); o += (size_t)Nn * 32;
  float*    z1   = (float*)(ws + o);    o += (size_t)Nn * 32;
  float*    a2s  = (float*)(ws + o);    o += (size_t)Nn * 32;
  float*    a2d  = (float*)(ws + o);    o += (size_t)Nn * 32;
  unsigned* m2   = (unsigned*)(ws + o); o += (size_t)Nn * 32;
  float*    z2   = (float*)(ws + o);    o += (size_t)Nn * 32;
  _Float16* h1h  = (_Float16*)(ws + o); o += (size_t)Nn * 128;   // N x 64 f16
  (void)ws_size;

  const int B = 256;
  const long long eTotH = (long long)(E + Nn) * H1;     // same for both layers (H1==H2)

  // ---- precision staging: f32 -> f16; weights packed into B-fragment lane order ----
  k_f32_to_f16<<<cdivll((long long)Nn * 256, B), B, 0, stream>>>(x, xh, (long long)Nn * 256);
  k_pack_B<256, 64><<<cdivll(4 * 8 * 32, B), B, 0, stream>>>(W1, W1p);
  k_pack_B<64, 128><<<cdivll(8 * 2 * 32, B), B, 0, stream>>>(W2, W2p);

  // ---- layer 1 ----
  {
    long long waves = (long long)(Nn / 16) * (F1 / 16);
    k_gemm_wmma<256, 64><<<cdivll(waves * 32, B), B, 0, stream>>>(xh, (const v16h*)W1p, xl1, Nn);
  }
  k_att_coeff<<<cdivll((long long)Nn * H1, B), B, 0, stream>>>(xl1, as1, ad1, a1s, a1d, m1, z1, Nn, H1, C1);
  k_zero_b32<<<cdivll((long long)Nn * F1, B), B, 0, stream>>>((unsigned*)agg1, (long long)Nn * F1);
  k_edge_max<<<cdivll(eTotH, B), B, 0, stream>>>(src, dst, E, Nn, H1, a1s, a1d, m1);
  k_edge_sum<<<cdivll(eTotH, B), B, 0, stream>>>(src, dst, E, Nn, H1, a1s, a1d, m1, z1);
  k_edge_agg<<<cdivll(eTotH, B), B, 0, stream>>>(src, dst, E, Nn, H1, C1, a1s, a1d, m1, z1, xl1, agg1);
  k_final1<<<cdivll((long long)Nn * F1, B), B, 0, stream>>>(agg1, b1, h1h, Nn, F1);

  // ---- layer 2 ----
  {
    long long waves = (long long)(Nn / 16) * (F2 / 16);
    k_gemm_wmma<64, 128><<<cdivll(waves * 32, B), B, 0, stream>>>(h1h, (const v16h*)W2p, xl2, Nn);
  }
  k_att_coeff<<<cdivll((long long)Nn * H2, B), B, 0, stream>>>(xl2, as2, ad2, a2s, a2d, m2, z2, Nn, H2, C2);
  k_zero_b32<<<cdivll((long long)Nn * F2, B), B, 0, stream>>>((unsigned*)agg2, (long long)Nn * F2);
  k_edge_max<<<cdivll(eTotH, B), B, 0, stream>>>(src, dst, E, Nn, H2, a2s, a2d, m2);
  k_edge_sum<<<cdivll(eTotH, B), B, 0, stream>>>(src, dst, E, Nn, H2, a2s, a2d, m2, z2);
  k_edge_agg<<<cdivll(eTotH, B), B, 0, stream>>>(src, dst, E, Nn, H2, C2, a2s, a2d, m2, z2, xl2, agg2);

  // ---- bias + log_softmax -> d_out ----
  k_logsoftmax128<<<cdivll((long long)Nn * 32, B), B, 0, stream>>>(agg2, b2, (float*)d_out, Nn);
}
